// ScaledDotProduct_14766097564106
// MI455X (gfx1250) — compile-verified, matrix operand order checked by reading, not run on hardware
//
#include <hip/hip_runtime.h>
#include <hip/hip_bf16.h>

// ---------------- problem constants ----------------
#define BATCH     4
#define SEQ       4096
#define DIM       1024
#define HD        64                   // head dim
#define BS        (BATCH * SEQ)        // 16384 rows
#define INV_SCALE 0.125f               // 1/sqrt(64)

typedef __attribute__((ext_vector_type(16))) _Float16     v16h;
typedef __attribute__((ext_vector_type(8)))  _Float16     v8h;
typedef __attribute__((ext_vector_type(8)))  float        v8f;
typedef __attribute__((ext_vector_type(4)))  float        v4f;
typedef __attribute__((ext_vector_type(4)))  unsigned int u32x4;
typedef __attribute__((ext_vector_type(8)))  int          i32x8;
typedef __attribute__((ext_vector_type(4)))  int          i32x4;

// ---------------- WMMA helper ----------------
// D = A(16x32 f16) * B(32x16 f16) + C(16x16 f32)
__device__ __forceinline__ v8f wmma32(v16h a, v16h b, v8f c) {
  return __builtin_amdgcn_wmma_f32_16x16x32_f16(
      /*neg_a=*/false, a, /*neg_b=*/false, b,
      /*c_mod=*/(short)0, c, /*reuse_a=*/false, /*reuse_b=*/false);
}

// 16-bit WMMA operand loader (per-lane half-K layout):
//   regs[0..7]  = p[0..7]     (K = kb..kb+7 for this lane half-group)
//   regs[8..15] = p[16..23]   (K = kb+16..kb+23)
// Caller passes p = rowBase + kb. Works for global and LDS pointers.
__device__ __forceinline__ v16h load_pair_h(const _Float16* p) {
  v8h lo = *(const v8h*)(p);
  v8h hi = *(const v8h*)(p + 16);
  v16h a;
#pragma unroll
  for (int i = 0; i < 8; ++i) { a[i] = lo[i]; a[8 + i] = hi[i]; }
  return a;
}

// Same layout, sourced from fp32 memory with convert-to-f16.
__device__ __forceinline__ v16h load_pair_f32(const float* p) {
  v4f x0 = *(const v4f*)(p);
  v4f x1 = *(const v4f*)(p + 4);
  v4f x2 = *(const v4f*)(p + 16);
  v4f x3 = *(const v4f*)(p + 20);
  v16h a;
#pragma unroll
  for (int i = 0; i < 4; ++i) {
    a[i]      = (_Float16)x0[i];
    a[4 + i]  = (_Float16)x1[i];
    a[8 + i]  = (_Float16)x2[i];
    a[12 + i] = (_Float16)x3[i];
  }
  return a;
}

// ---------------- Tensor Data Mover: 2D tile global -> LDS ----------------
// Builds a D# (group0 + group1 per CDNA5 ISA 8.3/8.4) and issues
// TENSOR_LOAD_TO_LDS. Issue from one wave only; wait with s_wait_tensorcnt.
__device__ __forceinline__ void tdm_load_2d(const void* lds_ptr, const void* gptr,
                                            unsigned data_size_log2,   // 0=1B 1=2B 2=4B
                                            unsigned tile_d0, unsigned tile_d1,
                                            unsigned tensor_d0, unsigned tensor_d1,
                                            unsigned stride0 /* elements */) {
  unsigned long long ga = (unsigned long long)gptr;
  unsigned lds_addr = (unsigned)(unsigned long long)lds_ptr;   // addr[31:0] = LDS offset
  u32x4 g0;
  g0[0] = 1u;                                            // count=1 (valid user D#)
  g0[1] = lds_addr;                                      // LDS byte address
  g0[2] = (unsigned)(ga & 0xffffffffu);                  // global addr lo
  g0[3] = (unsigned)((ga >> 32) & 0x01ffffffu) | (2u << 30);   // addr[56:32] | type=2
  i32x8 g1;
  g1[0] = (int)(data_size_log2 << 16);                   // mask=0 | data_size
  g1[1] = (int)((tensor_d0 & 0xffffu) << 16);            // abar=0 | tensor_dim0 lo16
  g1[2] = (int)((tensor_d0 >> 16) | ((tensor_d1 & 0xffffu) << 16));
  g1[3] = (int)((tensor_d1 >> 16) | (tile_d0 << 16));    // tensor_dim1 hi | tile_dim0
  g1[4] = (int)(tile_d1 & 0xffffu);                      // tile_dim1 | tile_dim2=0
  g1[5] = (int)stride0;                                  // tensor_dim0_stride lo32
  g1[6] = 0;                                             // stride0 hi | stride1 lo
  g1[7] = 0;                                             // stride1 hi
  i32x4 z4 = {0, 0, 0, 0};
#if defined(__clang_major__) && (__clang_major__ >= 23)
  i32x8 z8 = {0, 0, 0, 0, 0, 0, 0, 0};
  __builtin_amdgcn_tensor_load_to_lds(g0, g1, z4, z4, z8, 0);
#else
  __builtin_amdgcn_tensor_load_to_lds(g0, g1, z4, z4, 0);
#endif
}

// ---------------- kernel 0: W transpose+convert ----------------
// Wc[m][n][k] = (f16) W_m[k][n]   (column-major so B-operand loads are b128)
__global__ void wconv_kernel(const float* __restrict__ Wq,
                             const float* __restrict__ Wk,
                             const float* __restrict__ Wv,
                             _Float16* __restrict__ wc) {
  int idx = blockIdx.x * 256 + threadIdx.x;        // < 3*DIM*HD
  int m = idx / (DIM * HD);
  int r = idx - m * (DIM * HD);
  int k = r / HD;
  int n = r - k * HD;
  const float* W = (m == 0) ? Wq : ((m == 1) ? Wk : Wv);
  wc[(size_t)m * DIM * HD + (size_t)n * DIM + k] = (_Float16)W[r];
}

// ---------------- kernel 1: projection  xh = x @ W + b  (f16 out) ----------
// grid.x = BS/16 (M tiles), block = 128 (4 waves); wave w owns N-tile w*16.
// The 16x1024 f32 x-tile is DMA'd once into LDS by the TDM and shared by all
// 4 waves (4x global-traffic saving on the dominant input).
// transposed==0: out[row][n] row-major.  transposed!=0: out[b][n][s] (for vh^T).
__global__ __launch_bounds__(128) void proj_kernel(
    const float* __restrict__ x,        // [BS, DIM] f32
    const _Float16* __restrict__ wc,    // [HD, DIM] f16 (col-major W)
    const float* __restrict__ bias,     // [HD]
    _Float16* __restrict__ out,
    int transposed) {
  __shared__ __align__(16) float sx[16 * DIM];     // 64 KB x-tile

  const int lane   = threadIdx.x & 31;
  const int wave   = threadIdx.x >> 5;
  const int lanelo = lane & 15;
  const int hgrp   = lane >> 4;
  const int kb     = hgrp * 8;
  const int mtile  = blockIdx.x;

  if (wave == 0) {
    tdm_load_2d(sx, x + (size_t)mtile * 16 * DIM, /*f32*/2,
                /*tile*/16 * DIM, 1, /*tensor*/16 * DIM, 1, 16 * DIM);
    __builtin_amdgcn_s_wait_tensorcnt((short)0);
  }
  __syncthreads();

  const int ncol = wave * 16 + lanelo;
  const float*    xrow = sx + (size_t)lanelo * DIM;
  const _Float16* wrow = wc + (size_t)ncol * DIM;

  v8f acc = {};
#pragma unroll 4
  for (int k0 = 0; k0 < DIM; k0 += 32) {
    v16h a = load_pair_f32(xrow + k0 + kb);
    v16h b = load_pair_h (wrow + k0 + kb);
    acc = wmma32(a, b, acc);
  }

  const float bv   = bias[ncol];
  const int   rowb = mtile * 16 + hgrp * 8;        // first of this lane's 8 rows
  if (!transposed) {
#pragma unroll
    for (int j = 0; j < 8; ++j)
      out[(size_t)(rowb + j) * HD + ncol] = (_Float16)(acc[j] + bv);
  } else {
    // out[b][ncol][s] : 8 consecutive s values -> one b128 store
    const int b = rowb / SEQ;
    const int s = rowb - b * SEQ;
    v8h pk;
#pragma unroll
    for (int j = 0; j < 8; ++j) pk[j] = (_Float16)(acc[j] + bv);
    *(v8h*)(out + (size_t)b * SEQ * HD + (size_t)ncol * SEQ + s) = pk;
  }
}

// ---------------- kernel 2: sigmoid attention -------------------------------
// grid = (SEQ/64, BATCH), block = 128 (4 waves, one 16-query tile each).
// kh tiles (32x64 f16) are double-buffered in LDS via TDM (issued by wave 0,
// waited with s_wait_tensorcnt). vh is read directly from global in
// transposed [b][d][s] layout -> contiguous b128 B-operand loads.
__global__ __launch_bounds__(128) void attn_kernel(
    const _Float16* __restrict__ qh,   // [B, S, HD] f16
    const _Float16* __restrict__ kh,   // [B, S, HD] f16
    const _Float16* __restrict__ vhT,  // [B, HD, S] f16
    float* __restrict__ out) {         // [B, S, HD] f32
  __shared__ __align__(16) _Float16 skh[2][32 * HD];   // 2 x 4 KB double buffer
  __shared__ __align__(16) _Float16 ssc[4][16 * 40];   // per-wave score scratch

  const int lane   = threadIdx.x & 31;
  const int wave   = threadIdx.x >> 5;
  const int lanelo = lane & 15;
  const int hgrp   = lane >> 4;
  const int kb     = hgrp * 8;

  const int    b     = blockIdx.y;
  const int    qbase = (blockIdx.x * 4 + wave) * 16;
  const size_t bh    = (size_t)b * SEQ * HD;

  // Resident A operand: this wave's 16 query rows (K = d, 0..63).
  const _Float16* qrow = qh + bh + (size_t)(qbase + lanelo) * HD;
  const v16h aq0 = load_pair_h(qrow + kb);        // d 0..31
  const v16h aq1 = load_pair_h(qrow + 32 + kb);   // d 32..63

  v8f acc[4] = {{}, {}, {}, {}};
  _Float16* sc = ssc[wave];

  if (wave == 0)
    tdm_load_2d(skh[0], kh + bh, /*f16*/1, HD, 32, HD, SEQ, HD);

  for (int kt = 0; kt < SEQ; kt += 32) {
    const int cur = (kt >> 5) & 1;
    if (wave == 0) __builtin_amdgcn_s_wait_tensorcnt((short)0);
    __syncthreads();   // kh[cur] ready; all waves done reading kh[cur^1]
    if (wave == 0 && kt + 32 < SEQ)
      tdm_load_2d(skh[cur ^ 1], kh + bh + (size_t)(kt + 32) * HD,
                  1, HD, 32, HD, SEQ, HD);

    // ---- scores for two 16-key groups: z = sigmoid((qh . kh^T)/8) ----
    const _Float16* kt_lds = skh[cur];
#pragma unroll
    for (int g = 0; g < 2; ++g) {
      const _Float16* krow = kt_lds + (size_t)(g * 16 + lanelo) * HD;
      v16h bk0 = load_pair_h(krow + kb);        // d 0..31
      v16h bk1 = load_pair_h(krow + 32 + kb);   // d 32..63
      v8f c = {};
      c = wmma32(aq0, bk0, c);
      c = wmma32(aq1, bk1, c);
#pragma unroll
      for (int j = 0; j < 8; ++j) {
        float z = 1.0f / (1.0f + __expf(-c[j] * INV_SCALE));
        // C/D layout: row = j + 8*hgrp, col(key) = g*16 + lanelo
        sc[(j + 8 * hgrp) * 40 + g * 16 + lanelo] = (_Float16)z;
      }
    }

    // re-read scores in A layout: 16 q-rows x 32 keys
    v16h az = load_pair_h(sc + lanelo * 40 + kb);

    // ---- out += z @ vh : B operand straight from vh^T (contiguous in k) ----
#pragma unroll
    for (int dt = 0; dt < 4; ++dt) {
      const _Float16* vcol = vhT + bh + (size_t)(dt * 16 + lanelo) * SEQ + kt;
      v16h bv = load_pair_h(vcol + kb);
      acc[dt] = wmma32(az, bv, acc[dt]);
    }
  }

  // write out [16 x 64] f32 tile
#pragma unroll
  for (int dt = 0; dt < 4; ++dt) {
#pragma unroll
    for (int j = 0; j < 8; ++j) {
      int row = qbase + j + 8 * hgrp;
      int d   = dt * 16 + lanelo;
      out[bh + (size_t)row * HD + d] = acc[dt][j];
    }
  }
}

// ---------------- launcher ----------------
extern "C" void kernel_launch(void* const* d_in, const int* in_sizes, int n_in,
                              void* d_out, int out_size, void* d_ws, size_t ws_size,
                              hipStream_t stream) {
  const float* q  = (const float*)d_in[0];
  const float* k  = (const float*)d_in[1];
  const float* v  = (const float*)d_in[2];
  const float* Wq = (const float*)d_in[3];
  const float* bq = (const float*)d_in[4];
  const float* Wk = (const float*)d_in[5];
  const float* bk = (const float*)d_in[6];
  const float* Wv = (const float*)d_in[7];
  const float* bv = (const float*)d_in[8];
  float* out = (float*)d_out;

  // workspace layout (bytes):
  //   [0, 384K)  Wc: 3 x [HD, DIM] f16 (col-major transposed weights)
  //   then qh [B,S,HD], kh [B,S,HD], vhT [B,HD,S]: each BS*HD f16 = 2 MB
  char* ws = (char*)d_ws;
  _Float16* wc = (_Float16*)ws;
  const size_t WC_BYTES = (size_t)3 * DIM * HD * sizeof(_Float16);
  _Float16* qh  = (_Float16*)(ws + WC_BYTES);
  _Float16* kh  = qh + (size_t)BS * HD;
  _Float16* vhT = kh + (size_t)BS * HD;

  // 0) transpose/convert weights
  wconv_kernel<<<(3 * DIM * HD) / 256, 256, 0, stream>>>(Wq, Wk, Wv, wc);

  // 1) projections (f16 outputs; V written transposed)
  proj_kernel<<<BS / 16, 128, 0, stream>>>(q, wc,                     bq, qh,  0);
  proj_kernel<<<BS / 16, 128, 0, stream>>>(k, wc + (size_t)DIM * HD,  bk, kh,  0);
  proj_kernel<<<BS / 16, 128, 0, stream>>>(v, wc + (size_t)2*DIM*HD,  bv, vhT, 1);

  // 2) sigmoid attention
  dim3 grid(SEQ / 64, BATCH);
  attn_kernel<<<grid, 128, 0, stream>>>(qh, kh, vhT, out);
}
